// SelfAttLayer_9706626089920
// MI455X (gfx1250) — compile-verified, matrix operand order checked by reading, not run on hardware
//
#include <hip/hip_runtime.h>
#include <hip/hip_bf16.h>
#include <math.h>

typedef __attribute__((ext_vector_type(16))) __bf16 v16bf;
typedef __attribute__((ext_vector_type(8)))  __bf16 bf16x8;
typedef __attribute__((ext_vector_type(8)))  float  v8f;

// Problem constants (from reference)
constexpr int BB = 16;
constexpr int TT = 1500;
constexpr int DD = 512;
constexpr int HH = 1024;
constexpr int LL = 20;
constexpr int MM = BB * TT;   // 24000 rows

// ---------------------------------------------------------------------------
// Prep kernels: fp32 -> bf16 casts (+ transpose for the B operands)
// ---------------------------------------------------------------------------
__global__ void cast_f32_bf16(const float* __restrict__ in, __bf16* __restrict__ out, int n) {
    int i = blockIdx.x * blockDim.x + threadIdx.x;
    int stride = gridDim.x * blockDim.x;
    for (; i < n; i += stride) out[i] = (__bf16)in[i];
}

// in: [K][N] row-major fp32  ->  out: [N][K] row-major bf16 (i.e. B^T)
// K, N are compile-time powers of two -> index math is shifts, no division.
template <int K, int N>
__global__ void transpose_cast(const float* __restrict__ in, __bf16* __restrict__ out) {
    int i = blockIdx.x * blockDim.x + threadIdx.x;
    int stride = gridDim.x * blockDim.x;
    for (; i < K * N; i += stride) {
        int n = i / K;           // folds to shift
        int k = i - n * K;
        out[i] = (__bf16)in[(size_t)k * N + n];
    }
}

// ---------------------------------------------------------------------------
// bf16 WMMA GEMM: C[MM,N] = A[MM,K] * Bt[N,K]^T   (Bt row-major N x K)
// Block tile 128(M) x 128(N), K-step 32, double-buffered LDS with register
// prefetch. 8 waves: 4(M) x 2(N); each wave 32x64 = 2x4 wmma accumulators.
// Epilogue: RELU_BIAS_BF16 -> bf16(relu(c+bias)) staged via LDS for b128
// coalesced stores; else direct fp32 stores (immediate row offsets).
// ---------------------------------------------------------------------------
template <bool RELU_BIAS_BF16, int N, int K>
__global__ __launch_bounds__(256)
void gemm_bf16_wmma(const __bf16* __restrict__ A, const __bf16* __restrict__ Bt,
                    const float* __restrict__ bias, void* __restrict__ Cout) {
    constexpr int LDP = 40;                 // LDS tile row pitch (elements, 80B)
    constexpr int TILE_ELEMS = 128 * LDP;   // 5120 elements = 10240 B
    constexpr int CP = 136;                 // C-staging pitch (272B = 17 x 16B)
    constexpr int KT = K / 32;
    // 2 x (A+B) double buffer = 40960 B; C staging 128*136*2 = 34816 B (reused)
    __shared__ alignas(16) char smem[2 * 2 * TILE_ELEMS * 2];

    const int tid  = threadIdx.x;
    const int lane = tid & 31;
    const int wid  = tid >> 5;
    const int wm   = wid & 3;               // wave M quadrant: rows wm*32
    const int wn   = wid >> 2;              // wave N half:     cols wn*64

    const int blockN = blockIdx.x * 128;
    const int blockM = blockIdx.y * 128;

    const int mrow  = lane & 15;
    const int khalf = lane >> 4;

    // ---- per-thread staging geometry (2 x 16B chunks per matrix) ----
    size_t aOff[2], bOff[2];
    int    lOff[2];
#pragma unroll
    for (int s = 0; s < 2; ++s) {
        int chunk = tid + s * 256;          // 512 chunks of 16B per tile
        int row   = chunk >> 2;
        int c16   = chunk & 3;
        int gr    = blockM + row;
        if (gr >= MM) gr = MM - 1;          // clamp (stores are guarded)
        aOff[s] = (size_t)gr * K + c16 * 8;
        bOff[s] = (size_t)(blockN + row) * K + c16 * 8;
        lOff[s] = row * LDP + c16 * 8;
    }

    v8f acc[2][4];
#pragma unroll
    for (int r = 0; r < 2; ++r)
#pragma unroll
        for (int c = 0; c < 4; ++c)
            acc[r][c] = v8f{0.f, 0.f, 0.f, 0.f, 0.f, 0.f, 0.f, 0.f};

    // ---- prefetch K-tile 0 into registers ----
    bf16x8 ra[2], rb[2];
#pragma unroll
    for (int s = 0; s < 2; ++s) {
        ra[s] = *(const bf16x8*)(A  + aOff[s]);
        rb[s] = *(const bf16x8*)(Bt + bOff[s]);
    }

    int buf = 0;
    for (int kt = 0; kt < KT; ++kt) {
        __bf16* cA = (__bf16*)(smem + buf * (2 * TILE_ELEMS * 2));
        __bf16* cB = cA + TILE_ELEMS;

        // commit current tile to LDS
#pragma unroll
        for (int s = 0; s < 2; ++s) {
            *(bf16x8*)(&cA[lOff[s]]) = ra[s];
            *(bf16x8*)(&cB[lOff[s]]) = rb[s];
        }
        // issue next tile's global loads (complete under the WMMA burst)
        if (kt + 1 < KT) {
            int kb = (kt + 1) * 32;
#pragma unroll
            for (int s = 0; s < 2; ++s) {
                ra[s] = *(const bf16x8*)(A  + aOff[s] + kb);
                rb[s] = *(const bf16x8*)(Bt + bOff[s] + kb);
            }
        }
        __syncthreads();

        // fragments per ISA 16-bit 16x32 layout:
        // lanes 0-15: K 0..7 U 16..23 ; lanes 16-31: K 8..15 U 24..31
        v16bf afrag[2];
#pragma unroll
        for (int r = 0; r < 2; ++r) {
            int lrow = wm * 32 + r * 16 + mrow;
            bf16x8 lo = *(const bf16x8*)(&cA[lrow * LDP + khalf * 8]);
            bf16x8 hi = *(const bf16x8*)(&cA[lrow * LDP + 16 + khalf * 8]);
#pragma unroll
            for (int i = 0; i < 8; ++i) { afrag[r][i] = lo[i]; afrag[r][8 + i] = hi[i]; }
        }
        v16bf bfrag[4];
#pragma unroll
        for (int c = 0; c < 4; ++c) {
            int lrow = wn * 64 + c * 16 + mrow;
            bf16x8 lo = *(const bf16x8*)(&cB[lrow * LDP + khalf * 8]);
            bf16x8 hi = *(const bf16x8*)(&cB[lrow * LDP + 16 + khalf * 8]);
#pragma unroll
            for (int i = 0; i < 8; ++i) { bfrag[c][i] = lo[i]; bfrag[c][8 + i] = hi[i]; }
        }

#pragma unroll
        for (int r = 0; r < 2; ++r)
#pragma unroll
            for (int c = 0; c < 4; ++c)
                acc[r][c] = __builtin_amdgcn_wmma_f32_16x16x32_bf16(
                    false, afrag[r], false, bfrag[c], (short)0, acc[r][c], false, false);
        buf ^= 1;
    }

    const int nloc  = lane & 15;
    const int mhalf = (lane >> 4) * 8;

    if constexpr (RELU_BIAS_BF16) {
        // ---- bf16 epilogue: stage tile in LDS, then coalesced b128 stores ----
        __syncthreads();                     // done reading lA/lB; reuse as lC
        __bf16* lC = (__bf16*)smem;          // [128][CP]
#pragma unroll
        for (int r = 0; r < 2; ++r) {
#pragma unroll
            for (int c = 0; c < 4; ++c) {
                int nl = wn * 64 + c * 16 + nloc;
                float bv = bias[blockN + nl];
                int mb = wm * 32 + r * 16 + mhalf;
#pragma unroll
                for (int j = 0; j < 8; ++j) {
                    float v = acc[r][c][j] + bv;
                    v = v > 0.f ? v : 0.f;
                    lC[(mb + j) * CP + nl] = (__bf16)v;
                }
            }
        }
        __syncthreads();
        __bf16* Cb = (__bf16*)Cout;
#pragma unroll
        for (int s = 0; s < 8; ++s) {        // 2048 chunks of 16B
            int chunk = tid + s * 256;
            int row   = chunk >> 4;
            int col   = chunk & 15;
            int m     = blockM + row;
            if (m < MM) {
                bf16x8 v = *(const bf16x8*)(&lC[row * CP + col * 8]);
                *(bf16x8*)(Cb + (size_t)m * N + blockN + col * 8) = v;
            }
        }
    } else {
        // ---- fp32 epilogue: uniform full/partial branch, immediate offsets ----
        float* Cf = (float*)Cout;
        const bool full = (blockM + 128 <= MM);
#pragma unroll
        for (int r = 0; r < 2; ++r) {
#pragma unroll
            for (int c = 0; c < 4; ++c) {
                int n  = blockN + wn * 64 + c * 16 + nloc;
                int mb = blockM + wm * 32 + r * 16 + mhalf;
                float* base = Cf + (size_t)mb * N + n;
                if (full) {
#pragma unroll
                    for (int j = 0; j < 8; ++j) base[(size_t)j * N] = acc[r][c][j];
                } else {
#pragma unroll
                    for (int j = 0; j < 8; ++j)
                        if (mb + j < MM) base[(size_t)j * N] = acc[r][c][j];
                }
            }
        }
    }
}

// ---------------------------------------------------------------------------
// att = softmax(x @ wa, axis=-1): one wave32 per row, row staged in LDS,
// lane l (<20) owns one logit, shuffle reductions for max/sum.
// ---------------------------------------------------------------------------
__global__ __launch_bounds__(256)
void att_softmax(const float* __restrict__ x, const float* __restrict__ wa,
                 float* __restrict__ att, int Mrows) {
    __shared__ float xs[8][DD];
    const int w    = threadIdx.x >> 5;
    const int lane = threadIdx.x & 31;
    const int row  = blockIdx.x * 8 + w;
    if (row >= Mrows) return;

    const float4* xr  = (const float4*)(x + (size_t)row * DD);
    float4*       xsr = (float4*)xs[w];
#pragma unroll
    for (int i = 0; i < 4; ++i) xsr[lane + i * 32] = xr[lane + i * 32];
    __syncthreads();

    float dot = 0.f;
    if (lane < LL) {
#pragma unroll 4
        for (int k = 0; k < DD; ++k) dot += xs[w][k] * wa[k * LL + lane];
    }
    float v = (lane < LL) ? dot : -3.402823e38f;
#pragma unroll
    for (int off = 16; off > 0; off >>= 1) {
        float o = __shfl_xor(v, off, 32);
        v = v > o ? v : o;
    }
    float e = (lane < LL) ? __expf(dot - v) : 0.f;
    float s = e;
#pragma unroll
    for (int off = 16; off > 0; off >>= 1) s += __shfl_xor(s, off, 32);
    if (lane < LL) att[(size_t)row * LL + lane] = e / s;
}

// ---------------------------------------------------------------------------
// out[b,t,d] = x[b,t,d] + sum_{i<20, t-i>=0} p1[b,t-i,d] * att[b,t,i]
// One block per (b,t) row; 128 threads x float4 cover D=512.
// ---------------------------------------------------------------------------
__global__ __launch_bounds__(128)
void conv_mix(const float* __restrict__ x, const float* __restrict__ p1,
              const float* __restrict__ att, float* __restrict__ out) {
    __shared__ float a[LL];
    const int row = blockIdx.x;
    const int b   = row / TT;
    const int t   = row - b * TT;
    const int tid = threadIdx.x;

    if (tid < LL) a[tid] = att[(size_t)row * LL + tid];
    __syncthreads();

    float4 acc = ((const float4*)(x + (size_t)row * DD))[tid];
    for (int i = 0; i < LL; ++i) {
        int tt = t - i;
        if (tt < 0) break;
        float4 p = ((const float4*)(p1 + (size_t)(b * TT + tt) * DD))[tid];
        acc.x += a[i] * p.x;
        acc.y += a[i] * p.y;
        acc.z += a[i] * p.z;
        acc.w += a[i] * p.w;
    }
    ((float4*)(out + (size_t)row * DD))[tid] = acc;
}

// ---------------------------------------------------------------------------
extern "C" void kernel_launch(void* const* d_in, const int* in_sizes, int n_in,
                              void* d_out, int out_size, void* d_ws, size_t ws_size,
                              hipStream_t stream) {
    (void)in_sizes; (void)n_in; (void)out_size; (void)ws_size;
    const float* x  = (const float*)d_in[0];
    const float* w1 = (const float*)d_in[1];
    const float* b1 = (const float*)d_in[2];
    const float* wp = (const float*)d_in[3];
    const float* wa = (const float*)d_in[4];

    char*  ws  = (char*)d_ws;
    size_t off = 0;
    auto alloc = [&](size_t bytes) {
        void* p = ws + off;
        off += (bytes + 255) & ~size_t(255);
        return p;
    };
    __bf16* xbf = (__bf16*)alloc((size_t)MM * DD * 2);  // x in bf16
    __bf16* w1t = (__bf16*)alloc((size_t)HH * DD * 2);  // w1^T  [H][D]
    __bf16* wpt = (__bf16*)alloc((size_t)DD * HH * 2);  // wp^T  [D][H]
    __bf16* f1  = (__bf16*)alloc((size_t)MM * HH * 2);  // relu(x@w1+b1), bf16
    float*  p1  = (float*) alloc((size_t)MM * DD * 4);  // f1@wp, fp32
    float*  att = (float*) alloc((size_t)MM * LL * 4);  // softmax(x@wa)

    cast_f32_bf16<<<1024, 256, 0, stream>>>(x, xbf, MM * DD);
    transpose_cast<DD, HH><<<512, 256, 0, stream>>>(w1, w1t);
    transpose_cast<HH, DD><<<512, 256, 0, stream>>>(wp, wpt);
    att_softmax<<<MM / 8, 256, 0, stream>>>(x, wa, att, MM);

    dim3 g1(HH / 128, (MM + 127) / 128);
    gemm_bf16_wmma<true, HH, DD><<<g1, 256, 0, stream>>>(xbf, w1t, b1, (void*)f1);

    dim3 g2(DD / 128, (MM + 127) / 128);
    gemm_bf16_wmma<false, DD, HH><<<g2, 256, 0, stream>>>(f1, wpt, nullptr, (void*)p1);

    conv_mix<<<MM, 128, 0, stream>>>(x, p1, att, (float*)d_out);
}
